// simpleVIT_15788299780754
// MI455X (gfx1250) — compile-verified, matrix operand order checked by reading, not run on hardware
//
#include <hip/hip_runtime.h>
#include <hip/hip_bf16.h>
#include <math.h>

typedef __attribute__((ext_vector_type(16))) __bf16 v16bf;
typedef __attribute__((ext_vector_type(8)))  float  v8f;

#define EMB   128
#define NTOK  4096            // 64*64 tokens per batch image
#define BATCH 4
#define HEADS 4
#define DH    32
#define TOKENS (BATCH * NTOK) // 16384

static __device__ __forceinline__ v8f wmma_bf16(v16bf a, v16bf b, v8f c) {
  // D = A(16x32 bf16) * B(32x16 bf16) + C(16x16 f32)
  return __builtin_amdgcn_wmma_f32_16x16x32_bf16(false, a, false, b, (short)0, c,
                                                 false, false);
}

// ---------------------------------------------------------------------------
// 1) Patch embed: 5x5 same conv, 16 -> 128 channels. One block per output
//    pixel; the 16x5x5 input window is staged in LDS once and broadcast.
// ---------------------------------------------------------------------------
__global__ void conv_embed(const float* __restrict__ x, const float* __restrict__ w,
                           const float* __restrict__ bias, float* __restrict__ t) {
  __shared__ float win[400];
  const int tid = threadIdx.x;     // 0..127 = output channel
  const int bn  = blockIdx.x;      // b*4096 + n
  const int b   = bn >> 12;
  const int n   = bn & 4095;
  const int py  = n >> 6;
  const int px  = n & 63;
  for (int idx = tid; idx < 400; idx += 128) {
    const int ci = idx / 25, k = idx % 25;
    const int dy = k / 5, dx = k % 5;
    const int yy = py + dy - 2, xx = px + dx - 2;
    float v = 0.0f;
    if ((unsigned)yy < 64u && (unsigned)xx < 64u)
      v = x[(((b * 16 + ci) * 64 + yy) << 6) + xx];
    win[idx] = v;
  }
  __syncthreads();
  float acc = bias[tid];
  const float* wp = w + (size_t)tid * 400;
  #pragma unroll 8
  for (int i = 0; i < 400; ++i) acc = fmaf(wp[i], win[i], acc);
  t[(size_t)bn * EMB + tid] = acc;
}

// ---------------------------------------------------------------------------
// 2) LayerNorm (E=128) -> bf16. One wave per token, 4 elements per lane,
//    wave32 shfl_xor reductions.
// ---------------------------------------------------------------------------
__global__ void layernorm_to_bf16(const float* __restrict__ in, const float* __restrict__ g,
                                  const float* __restrict__ bta, __bf16* __restrict__ out) {
  const int lane = threadIdx.x & 31;
  const int wave = threadIdx.x >> 5;
  const int tok  = blockIdx.x * 8 + wave;
  const float4 xv = *(const float4*)(in + (size_t)tok * EMB + lane * 4);
  float s = xv.x + xv.y + xv.z + xv.w;
  #pragma unroll
  for (int off = 1; off < 32; off <<= 1) s += __shfl_xor(s, off, 32);
  const float mu = s * (1.0f / 128.0f);
  const float d0 = xv.x - mu, d1 = xv.y - mu, d2 = xv.z - mu, d3 = xv.w - mu;
  float q = d0 * d0 + d1 * d1 + d2 * d2 + d3 * d3;
  #pragma unroll
  for (int off = 1; off < 32; off <<= 1) q += __shfl_xor(q, off, 32);
  const float rs = rsqrtf(q * (1.0f / 128.0f) + 1e-5f);
  const float4 gv = *(const float4*)(g + lane * 4);
  const float4 bv = *(const float4*)(bta + lane * 4);
  __bf16* op = out + (size_t)tok * EMB + lane * 4;
  op[0] = (__bf16)(d0 * rs * gv.x + bv.x);
  op[1] = (__bf16)(d1 * rs * gv.y + bv.y);
  op[2] = (__bf16)(d2 * rs * gv.z + bv.z);
  op[3] = (__bf16)(d3 * rs * gv.w + bv.w);
}

__global__ void f32_to_bf16(const float* __restrict__ in, __bf16* __restrict__ out, int n) {
  const int i = blockIdx.x * blockDim.x + threadIdx.x;
  if (i < n) out[i] = (__bf16)in[i];
}

// ---------------------------------------------------------------------------
// 3) QKV projection: C[M,128] = Y[M,128] @ W^T + b  (C = A*B^T, both row-major
//    -> contiguous per-lane WMMA fragments). One 16x16 tile per wave, 8 waves
//    cover all 8 column tiles; K=128 = 4 x wmma_16x16x32.
//    Q,K stored [bh, N, 32]; V stored transposed [bh, 32, N] for P@V.
// ---------------------------------------------------------------------------
__global__ void qkv_gemm(const __bf16* __restrict__ Y,
                         const __bf16* __restrict__ Wq, const __bf16* __restrict__ Wk,
                         const __bf16* __restrict__ Wv,
                         const float* __restrict__ Bq, const float* __restrict__ Bk,
                         const float* __restrict__ Bv,
                         __bf16* __restrict__ Qo, __bf16* __restrict__ Ko,
                         __bf16* __restrict__ VTo) {
  const int lane = threadIdx.x & 31;
  const int wave = threadIdx.x >> 5;
  const int m0 = blockIdx.x * 16;
  const int n0 = wave * 16;
  const int which = blockIdx.y;
  const __bf16* W = (which == 0) ? Wq : ((which == 1) ? Wk : Wv);
  const float*  B = (which == 0) ? Bq : ((which == 1) ? Bk : Bv);
  const int hl = lane & 15, hh = lane >> 4;
  const __bf16* ap = Y + (size_t)(m0 + hl) * EMB + hh * 16;
  const __bf16* bp = W + (size_t)(n0 + hl) * EMB + hh * 16;
  v8f c = {};
  #pragma unroll
  for (int kk = 0; kk < 4; ++kk) {
    const v16bf a = *(const v16bf*)(ap + kk * 32);
    const v16bf b = *(const v16bf*)(bp + kk * 32);
    c = wmma_bf16(a, b, c);
  }
  const int col = n0 + hl;
  const float bias = B[col];
  const int h = col >> 5, d = col & 31;
  #pragma unroll
  for (int r = 0; r < 8; ++r) {
    const int m = m0 + r + (hh << 3);
    const int bb = m >> 12, nn = m & 4095;
    const int bh = bb * HEADS + h;
    const float v = c[r] + bias;
    if (which == 0)      Qo[((size_t)bh * NTOK + nn) * DH + d] = (__bf16)v;
    else if (which == 1) Ko[((size_t)bh * NTOK + nn) * DH + d] = (__bf16)v;
    else                 VTo[((size_t)bh * DH + d) * NTOK + nn] = (__bf16)v;
  }
}

// ---------------------------------------------------------------------------
// 4) Flash attention per (b,h). Each wave owns 16 query rows, streams keys in
//    tiles of 32. Per tile: 2 WMMAs for S, 2 WMMAs for P@V, and 1 WMMA with an
//    all-ones B fragment that accumulates the softmax denominator (row-sum of
//    P lands in every lane's column -> no cross-lane sum reduction needed).
//    Only the running row-max still uses shfl reductions.
//    Reference semantics: softmax first, then divide by sqrt(EMB) -> fold a
//    single 1/sqrt(128) into the final normalization.
// ---------------------------------------------------------------------------
__global__ void flash_attn(const __bf16* __restrict__ Qb, const __bf16* __restrict__ Kb,
                           const __bf16* __restrict__ VTb, __bf16* __restrict__ Ob) {
  __shared__ __align__(64) __bf16 plds[8][16 * 32];
  const int lane = threadIdx.x & 31;
  const int wave = threadIdx.x >> 5;
  const int bh = blockIdx.x;                       // b*HEADS + h
  const int q0 = blockIdx.y * 128 + wave * 16;
  const int hl = lane & 15, hh = lane >> 4;
  const __bf16* Q  = Qb  + (size_t)bh * NTOK * DH;
  const __bf16* Kp = Kb  + (size_t)bh * NTOK * DH;
  const __bf16* VT = VTb + (size_t)bh * DH * NTOK;
  const v16bf qf = *(const v16bf*)(Q + (size_t)(q0 + hl) * DH + hh * 16);
  v16bf onesf;
  #pragma unroll
  for (int i = 0; i < 16; ++i) onesf[i] = (__bf16)1.0f;
  v8f o0 = {}, o1 = {}, o2 = {};       // o2 = running row-sum of P (all cols)
  float mrow[8];
  #pragma unroll
  for (int r = 0; r < 8; ++r) mrow[r] = -3.0e38f;
  __bf16* pt = &plds[wave][0];

  for (int j = 0; j < NTOK; j += 32) {
    const v16bf kf0 = *(const v16bf*)(Kp + (size_t)(j + hl) * DH + hh * 16);
    const v16bf kf1 = *(const v16bf*)(Kp + (size_t)(j + 16 + hl) * DH + hh * 16);
    if (j + 32 < NTOK) {               // gfx1250 global_prefetch for next tiles
      __builtin_prefetch(Kp + (size_t)(j + 32 + hl) * DH, 0, 1);
      __builtin_prefetch(VT + (size_t)hl * NTOK + (j + 32), 0, 1);
    }
    const v8f zacc = {};
    const v8f s0 = wmma_bf16(qf, kf0, zacc);   // S[:,0:16]
    const v8f s1 = wmma_bf16(qf, kf1, zacc);   // S[:,16:32]

    #pragma unroll
    for (int r = 0; r < 8; ++r) {
      // C-layout: VGPR r holds row (r + 8*hh), column = hl, across 16 lanes.
      float mx = fmaxf(s0[r], s1[r]);
      #pragma unroll
      for (int off = 1; off < 16; off <<= 1) mx = fmaxf(mx, __shfl_xor(mx, off, 32));
      const float mn = fmaxf(mrow[r], mx);
      const float sc = __expf(mrow[r] - mn);
      mrow[r] = mn;
      const float p0 = __expf(s0[r] - mn);
      const float p1 = __expf(s1[r] - mn);
      o0[r] *= sc;
      o1[r] *= sc;
      o2[r] *= sc;                               // rescale denominator too
      const int row = r + (hh << 3);
      pt[row * 32 + hl]      = (__bf16)p0;       // P row-major 16x32 in LDS
      pt[row * 32 + 16 + hl] = (__bf16)p1;
    }
    __builtin_amdgcn_wave_barrier();
    asm volatile("s_wait_dscnt 0" ::: "memory");   // wave-private LDS WAR/RAW fence
    __builtin_amdgcn_wave_barrier();
    const v16bf pf = *(const v16bf*)(pt + hl * 32 + hh * 16);   // A-fragment of P
    const v16bf v0 = *(const v16bf*)(VT + (size_t)hl * NTOK + j + hh * 16);        // d 0..15
    const v16bf v1 = *(const v16bf*)(VT + (size_t)(16 + hl) * NTOK + j + hh * 16); // d 16..31
    o0 = wmma_bf16(pf, v0, o0);
    o1 = wmma_bf16(pf, v1, o1);
    o2 = wmma_bf16(pf, onesf, o2);   // denominator on the matrix pipe
    __builtin_amdgcn_wave_barrier();
  }

  const int bb = bh >> 2, h = bh & 3;
  const float inv = 0.08838834764831845f;  // 1/sqrt(128), applied after softmax
  #pragma unroll
  for (int r = 0; r < 8; ++r) {
    const int row = q0 + r + (hh << 3);
    const float rl = inv / o2[r];            // every lane holds its row's sum
    __bf16* op = Ob + ((size_t)(bb * NTOK + row)) * EMB + h * DH;
    op[hl]      = (__bf16)(o0[r] * rl);
    op[16 + hl] = (__bf16)(o1[r] * rl);
  }
}

// ---------------------------------------------------------------------------
// 5) Generic 128-wide WMMA GEMM with epilogues:
//    EPI 0: + bias + residual -> f32           (attention output projection)
//    EPI 1: + bias, exact GELU -> bf16         (MLP up)
//    EPI 2: + bias + residual -> f32 stored transposed [B,E,H*W] (final out)
// ---------------------------------------------------------------------------
template <int EPI>
__global__ void gemm128(const __bf16* __restrict__ A, const __bf16* __restrict__ W,
                        const float* __restrict__ B, const float* __restrict__ R,
                        float* __restrict__ Of, __bf16* __restrict__ Ob) {
  const int lane = threadIdx.x & 31;
  const int wave = threadIdx.x >> 5;
  const int m0 = blockIdx.x * 16;
  const int n0 = wave * 16;
  const int hl = lane & 15, hh = lane >> 4;
  const __bf16* ap = A + (size_t)(m0 + hl) * EMB + hh * 16;
  const __bf16* bp = W + (size_t)(n0 + hl) * EMB + hh * 16;
  v8f c = {};
  #pragma unroll
  for (int kk = 0; kk < 4; ++kk) {
    const v16bf a = *(const v16bf*)(ap + kk * 32);
    const v16bf b = *(const v16bf*)(bp + kk * 32);
    c = wmma_bf16(a, b, c);
  }
  const int col = n0 + hl;
  const float bias = B[col];
  #pragma unroll
  for (int r = 0; r < 8; ++r) {
    const int m = m0 + r + (hh << 3);
    float v = c[r] + bias;
    if (EPI == 0) {
      v += R[(size_t)m * EMB + col];
      Of[(size_t)m * EMB + col] = v;
    } else if (EPI == 1) {
      v = 0.5f * v * (1.0f + erff(v * 0.70710678118654752f));
      Ob[(size_t)m * EMB + col] = (__bf16)v;
    } else {
      v += R[(size_t)m * EMB + col];
      const int bb = m >> 12, nn = m & 4095;
      Of[((size_t)(bb * EMB + col)) * NTOK + nn] = v;
    }
  }
}

// ---------------------------------------------------------------------------
extern "C" void kernel_launch(void* const* d_in, const int* in_sizes, int n_in,
                              void* d_out, int out_size, void* d_ws, size_t ws_size,
                              hipStream_t stream) {
  (void)in_sizes; (void)n_in; (void)out_size; (void)ws_size;
  const float* x      = (const float*)d_in[0];
  const float* conv_w = (const float*)d_in[1];
  const float* conv_b = (const float*)d_in[2];
  const float* ln1_g  = (const float*)d_in[3];
  const float* ln1_b  = (const float*)d_in[4];
  const float* wq = (const float*)d_in[5];
  const float* wk = (const float*)d_in[6];
  const float* wv = (const float*)d_in[7];
  const float* wo = (const float*)d_in[8];
  const float* bq = (const float*)d_in[9];
  const float* bk = (const float*)d_in[10];
  const float* bv = (const float*)d_in[11];
  const float* bo = (const float*)d_in[12];
  const float* ffg = (const float*)d_in[13];
  const float* ffb = (const float*)d_in[14];
  const float* w1 = (const float*)d_in[15];
  const float* b1 = (const float*)d_in[16];
  const float* w2 = (const float*)d_in[17];
  const float* b2 = (const float*)d_in[18];

  char* ws = (char*)d_ws;
  float*  t    = (float*) (ws);                   // [16384,128] f32  (8 MB)
  float*  t2   = (float*) (ws + (8u  << 20));     // [16384,128] f32  (8 MB)
  __bf16* yln  = (__bf16*)(ws + (16u << 20));     // LN1 out, reused as LN2 out
  __bf16* qb   = (__bf16*)(ws + (20u << 20));     // Q [16,4096,32], reused as z1
  __bf16* kb   = (__bf16*)(ws + (24u << 20));     // K [16,4096,32]
  __bf16* vT   = (__bf16*)(ws + (28u << 20));     // V^T [16,32,4096]
  __bf16* aout = (__bf16*)(ws + (32u << 20));     // attn out [16384,128]
  __bf16* wqb  = (__bf16*)(ws + (36u << 20));     // bf16 weights, 6 x 16384
  __bf16* wkb  = wqb + 16384;
  __bf16* wvb  = wkb + 16384;
  __bf16* wob  = wvb + 16384;
  __bf16* w1b  = wob + 16384;
  __bf16* w2b  = w1b + 16384;
  __bf16* zln  = yln;   // overlay: yln dead after QKV
  __bf16* z1   = qb;    // overlay: Q dead after attention

  // Patch embed -> tokens t [B*N, E]
  conv_embed<<<TOKENS, 128, 0, stream>>>(x, conv_w, conv_b, t);

  // Weights f32 -> bf16
  f32_to_bf16<<<64, 256, 0, stream>>>(wq, wqb, 16384);
  f32_to_bf16<<<64, 256, 0, stream>>>(wk, wkb, 16384);
  f32_to_bf16<<<64, 256, 0, stream>>>(wv, wvb, 16384);
  f32_to_bf16<<<64, 256, 0, stream>>>(wo, wob, 16384);
  f32_to_bf16<<<64, 256, 0, stream>>>(w1, w1b, 16384);
  f32_to_bf16<<<64, 256, 0, stream>>>(w2, w2b, 16384);

  // Attention block
  layernorm_to_bf16<<<TOKENS / 8, 256, 0, stream>>>(t, ln1_g, ln1_b, yln);
  qkv_gemm<<<dim3(TOKENS / 16, 3), 256, 0, stream>>>(yln, wqb, wkb, wvb,
                                                     bq, bk, bv, qb, kb, vT);
  flash_attn<<<dim3(BATCH * HEADS, NTOK / 128), 256, 0, stream>>>(qb, kb, vT, aout);
  gemm128<0><<<TOKENS / 16, 256, 0, stream>>>(aout, wob, bo, t, t2, nullptr);

  // Feedforward block
  layernorm_to_bf16<<<TOKENS / 8, 256, 0, stream>>>(t2, ffg, ffb, zln);
  gemm128<1><<<TOKENS / 16, 256, 0, stream>>>(zln, w1b, b1, nullptr, nullptr, z1);
  gemm128<2><<<TOKENS / 16, 256, 0, stream>>>(z1, w2b, b2, t2, (float*)d_out, nullptr);
}
